// EGNN_Conv_Model_43843026157982
// MI455X (gfx1250) — compile-verified, matrix operand order checked by reading, not run on hardware
//
#include <hip/hip_runtime.h>

#define EPS_BN 1e-5f

typedef __attribute__((ext_vector_type(16))) __bf16 v16bf;
typedef __attribute__((ext_vector_type(8)))  float  v8f;

// float -> bf16: native conversion (gfx1250 has bf16 VALU; lowers to v_cvt_*_bf16_f32,
// pairs of casts can fuse into v_cvt_pk_bf16_f32)
__device__ __forceinline__ __bf16 f2bf(float f) {
  return (__bf16)f;
}

// K index for chunk c, VGPR j, element tt, K-half kh (ISA 16-bit A/B layout).
__device__ __forceinline__ constexpr int kOf(int c, int j, int tt, int kh) {
  return 32 * c + (j & 3) * 2 + ((j >> 2) << 4) + kh * 8 + tt;
}
// LDS delta for padded layout (stride 12, data at +2): index = base + ci*12 + kk
// where base = sl*CIN*12 + l.  Returns -1 if k is in the zero K-padding.
__device__ __forceinline__ constexpr int deltaOf(int CIN, int c, int j, int tt, int kh) {
  int k = kOf(c, j, tt, kh);
  if (k >= CIN * 5) return -1;
  int ci = k / 5, kk = k - ci * 5;
  return ci * 12 + kk;
}

// ---------------------------------------------------------------------------
// Conv1d(K=5,pad=2) resblock stage as WMMA GEMM over bf16 im2col.
//   MODE 0: dense [Ns, CIN, 8] (optional BN+ReLU on load)
//   MODE 1: edge gather concat(h[gidx0], h[gidx1])  (CIN == 16)
//   MODE 2: node concat(nodeA, nodeB)               (CIN == 16)
// M rows 0-7: main conv -> tOut (+stats). M rows 8-15: 1x1 skip -> skipOut.
// ---------------------------------------------------------------------------
template<int CIN, int MODE, bool INBN, bool SKIP>
__global__ void __launch_bounds__(256)
conv_rb_kernel(const float* Xin, const float* nodeA, const float* nodeB,
               const int* __restrict__ gidx0, const int* __restrict__ gidx1,
               const float* __restrict__ wmain, const float* __restrict__ bmain,
               const float* __restrict__ wskip, const float* __restrict__ bskip,
               const float* __restrict__ statsIn, const float* __restrict__ gIn,
               const float* __restrict__ beIn, float invInCnt,
               float* tOut, float* skipOut, float* __restrict__ statsOut, int Ns)
{
  constexpr int KREAL  = CIN * 5;
  constexpr int CHUNKS = (KREAL + 31) / 32;
  constexpr int ROWS   = 2 * CIN;          // rows staged per tile (2 samples)
  constexpr int NF4    = ROWS * 2;         // float4 chunks per tile (8 floats/row)
  constexpr int F4IT   = NF4 / 32;         // staging iterations (2 for CIN16, 1 for CIN8)

  const int lane = threadIdx.x & 31;
  const int m    = lane & 15;
  const int kh   = lane >> 4;

  // ---- A matrix (weights, loop-invariant) ----
  v16bf A[CHUNKS];
#pragma unroll
  for (int c = 0; c < CHUNKS; ++c)
#pragma unroll
    for (int j = 0; j < 8; ++j)
#pragma unroll
      for (int t = 0; t < 2; ++t) {
        int k = kOf(c, j, t, kh);
        float w = 0.f;
        if (k < KREAL) {
          int ci = k / 5, kk = k - ci * 5;
          if (m < 8)                w = wmain[(m * CIN + ci) * 5 + kk];
          else if (SKIP && kk == 2) w = wskip[(m - 8) * CIN + ci];
        }
        A[c][2 * j + t] = f2bf(w);
      }

  // ---- biases hoisted into registers ----
  float bR[8], bS[8];
#pragma unroll
  for (int r = 0; r < 8; ++r) {
    bR[r] = bmain[r];
    bS[r] = SKIP ? bskip[r] : 0.f;
  }

  float bnMean[8], bnScale[8], bnBias[8];
  if (INBN) {
#pragma unroll
    for (int c = 0; c < 8; ++c) {
      float mu  = statsIn[c] * invInCnt;
      float var = statsIn[8 + c] * invInCnt - mu * mu;
      bnMean[c]  = mu;
      bnScale[c] = rsqrtf(var + EPS_BN) * gIn[c];
      bnBias[c]  = beIn[c];
    }
  }

  __shared__ float stageAll[8 * ROWS * 12];
  __shared__ float trAll[8 * 256];
  float* sm = stageAll + (threadIdx.x >> 5) * (ROWS * 12);
  float* tr = trAll + (threadIdx.x >> 5) * 256;

  // zero the LDS pad slots once (they are never overwritten by data)
  for (int p2 = lane; p2 < ROWS * 4; p2 += 32) {
    int row = p2 >> 2, w = p2 & 3;
    sm[row * 12 + ((w < 2) ? w : w + 8)] = 0.f;
  }

  const int wave   = (blockIdx.x * blockDim.x + threadIdx.x) >> 5;
  const int nWaves = (gridDim.x * blockDim.x) >> 5;
  const int nTiles = (Ns + 1) >> 1;

  float ls[8], lq[8];
#pragma unroll
  for (int r = 0; r < 8; ++r) { ls[r] = 0.f; lq[r] = 0.f; }

  const int n    = lane & 15;
  const int sl   = n >> 3;
  const int l    = n & 7;
  const int base = sl * CIN * 12 + l;

  for (int t = wave; t < nTiles; t += nWaves) {
    const int s0 = t * 2;
    const bool fullTile = (s0 + 2 <= Ns);

    // ---- stage 2 samples into padded LDS: float4 loads, clamped (branchless) ----
#pragma unroll
    for (int it = 0; it < F4IT; ++it) {
      int flat4 = it * 32 + lane;                 // float4 index in tile
      int ss    = flat4 / (CIN * 2);
      int restf = flat4 - ss * (CIN * 2);
      int ci    = restf >> 1;
      int li    = (restf & 1) * 4;
      int s     = s0 + ss; s = (s < Ns) ? s : (Ns - 1);
      float4 v;
      if (MODE == 0) {
        v = *(const float4*)(Xin + (size_t)s * (CIN * 8) + restf * 4);
      } else if (MODE == 1) {
        int node = (ci < 8) ? gidx0[s] : gidx1[s];
        v = *(const float4*)(nodeA + (size_t)node * 64 + (ci & 7) * 8 + li);
      } else {
        const float* src = (ci < 8) ? nodeA : nodeB;
        v = *(const float4*)(src + (size_t)s * 64 + (ci & 7) * 8 + li);
      }
      if (INBN) {
        int c = ci & 7;
        v.x = fmaxf((v.x - bnMean[c]) * bnScale[c] + bnBias[c], 0.f);
        v.y = fmaxf((v.y - bnMean[c]) * bnScale[c] + bnBias[c], 0.f);
        v.z = fmaxf((v.z - bnMean[c]) * bnScale[c] + bnBias[c], 0.f);
        v.w = fmaxf((v.w - bnMean[c]) * bnScale[c] + bnBias[c], 0.f);
      }
      float* dst = sm + (ss * CIN + ci) * 12 + 2 + li;
      *(float2*)(dst)     = float2{v.x, v.y};
      *(float2*)(dst + 2) = float2{v.z, v.w};
    }

    // ---- B build (constant-folded LDS deltas) + WMMA accumulate ----
    v8f acc;
#pragma unroll
    for (int r = 0; r < 8; ++r) acc[r] = 0.f;
#pragma unroll
    for (int c = 0; c < CHUNKS; ++c) {
      v16bf B;
#pragma unroll
      for (int j = 0; j < 8; ++j)
#pragma unroll
        for (int tt = 0; tt < 2; ++tt) {
          const int d0 = deltaOf(CIN, c, j, tt, 0);
          const int d1 = deltaOf(CIN, c, j, tt, 1);
          float v;
          if (d0 < 0 && d1 < 0) {
            v = 0.f;                                   // compile-time zero pad
          } else {
            int dd = kh ? d1 : d0;
            float x = sm[base + (dd < 0 ? 0 : dd)];
            v = (dd < 0) ? 0.f : x;
          }
          B[2 * j + tt] = f2bf(v);
        }
      acc = __builtin_amdgcn_wmma_f32_16x16x32_bf16(false, A[c], false, B,
                                                    (short)0, acc, false, false);
    }

    // ---- epilogue: transpose through LDS, coalesced b128 stores ----
    if (kh == 0) {
      const bool valid = (s0 + sl) < Ns;
#pragma unroll
      for (int r = 0; r < 8; ++r) {
        float y = acc[r] + bR[r];
        tr[sl * 64 + r * 8 + l] = y;
        if (valid) { ls[r] += y; lq[r] += y * y; }
      }
    } else if (SKIP) {
#pragma unroll
      for (int r = 0; r < 8; ++r)
        tr[128 + sl * 64 + r * 8 + l] = acc[r] + bS[r];
    }

    {
      // conv rows: 128 floats at tr[0..), skip rows: 128 floats at tr[128..)
      const int flat  = lane * 4;
      const int bound = (Ns - s0) * 64;
      float4 v0 = *(const float4*)(tr + flat);
      if (fullTile || flat < bound)
        *(float4*)(tOut + (size_t)s0 * 64 + flat) = v0;
      if (SKIP) {
        float4 v1 = *(const float4*)(tr + 128 + flat);
        if (fullTile || flat < bound)
          *(float4*)(skipOut + (size_t)s0 * 64 + flat) = v1;
      }
    }
  }

  // ---- per-wave stats reduction, one atomic set per wave ----
#pragma unroll
  for (int r = 0; r < 8; ++r) {
    float s = ls[r], q = lq[r];
#pragma unroll
    for (int off = 16; off >= 1; off >>= 1) {
      s += __shfl_xor(s, off, 32);
      q += __shfl_xor(q, off, 32);
    }
    if (lane == 0) {
      atomicAdd(&statsOut[r], s);
      atomicAdd(&statsOut[8 + r], q);
    }
  }
}

// ---------------------------------------------------------------------------
// Combine: out = relu(bn(t2)) + skip  (float4), optional scatter-add.
// ---------------------------------------------------------------------------
__global__ void combine_kernel(const float* __restrict__ t2, const float* __restrict__ skip,
                               const float* __restrict__ stats, const float* __restrict__ g,
                               const float* __restrict__ be, float invCnt,
                               float* out, const int* __restrict__ sidx, float* agg, int Ns)
{
  long long i4 = (long long)blockIdx.x * blockDim.x + threadIdx.x;
  if (i4 >= (long long)Ns * 16) return;
  long long i = i4 * 4;
  int c = (int)((i >> 3) & 7);
  float mu  = stats[c] * invCnt;
  float var = stats[8 + c] * invCnt - mu * mu;
  float sc  = rsqrtf(var + EPS_BN) * g[c];
  float bb  = be[c];
  float4 tv = *(const float4*)(t2 + i);
  float4 sv = *(const float4*)(skip + i);
  float4 r;
  r.x = fmaxf((tv.x - mu) * sc + bb, 0.f) + sv.x;
  r.y = fmaxf((tv.y - mu) * sc + bb, 0.f) + sv.y;
  r.z = fmaxf((tv.z - mu) * sc + bb, 0.f) + sv.z;
  r.w = fmaxf((tv.w - mu) * sc + bb, 0.f) + sv.w;
  if (sidx) {
    float* dst = agg + (size_t)sidx[i >> 6] * 64 + (i & 63);
    atomicAdd(dst + 0, r.x); atomicAdd(dst + 1, r.y);
    atomicAdd(dst + 2, r.z); atomicAdd(dst + 3, r.w);
  } else {
    *(float4*)(out + i) = r;
  }
}

// ---------------------------------------------------------------------------
// FC1: z[n,f] = hf[n,:] @ W[:,f] + b.  M=16 FC units, K=64, WMMA bf16.
// ---------------------------------------------------------------------------
__global__ void __launch_bounds__(256)
fc1_wmma_kernel(const float* __restrict__ hf, const float* __restrict__ w,
                const float* __restrict__ b, float* __restrict__ z,
                float* __restrict__ statsOut, int Ns)
{
  const int lane = threadIdx.x & 31;
  const int m    = lane & 15;
  const int kh   = lane >> 4;

  v16bf A[2];
#pragma unroll
  for (int c = 0; c < 2; ++c)
#pragma unroll
    for (int j = 0; j < 8; ++j)
#pragma unroll
      for (int t = 0; t < 2; ++t) {
        int k = kOf(c, j, t, kh);              // k in [0,64)
        A[c][2 * j + t] = f2bf(w[k * 16 + m]); // A[f,d] = W[d,f]
      }

  const int f0 = kh * 8;
  float bv[8];
#pragma unroll
  for (int r = 0; r < 8; ++r) bv[r] = b[f0 + r];

  const int wave   = (blockIdx.x * blockDim.x + threadIdx.x) >> 5;
  const int nWaves = (gridDim.x * blockDim.x) >> 5;
  const int nTiles = (Ns + 15) >> 4;

  float ls[8], lq[8];
#pragma unroll
  for (int r = 0; r < 8; ++r) { ls[r] = 0.f; lq[r] = 0.f; }
  const int n = lane & 15;

  for (int t = wave; t < nTiles; t += nWaves) {
    const int  s     = t * 16 + n;
    const bool valid = s < Ns;
    const int  scl   = valid ? s : (Ns - 1);
    const float* row = hf + (size_t)scl * 64 + kh * 8;

    // four contiguous 8-float K-runs per lane (branchless, clamped)
    float rv[4][8];
#pragma unroll
    for (int rr = 0; rr < 4; ++rr) {
      float4 a0 = *(const float4*)(row + rr * 16);
      float4 a1 = *(const float4*)(row + rr * 16 + 4);
      rv[rr][0] = a0.x; rv[rr][1] = a0.y; rv[rr][2] = a0.z; rv[rr][3] = a0.w;
      rv[rr][4] = a1.x; rv[rr][5] = a1.y; rv[rr][6] = a1.z; rv[rr][7] = a1.w;
    }

    v8f acc;
#pragma unroll
    for (int r = 0; r < 8; ++r) acc[r] = 0.f;
#pragma unroll
    for (int c = 0; c < 2; ++c) {
      v16bf B;
#pragma unroll
      for (int j = 0; j < 8; ++j)
#pragma unroll
        for (int tt = 0; tt < 2; ++tt)
          B[2 * j + tt] = f2bf(rv[2 * c + (j >> 2)][(j & 3) * 2 + tt]);
      acc = __builtin_amdgcn_wmma_f32_16x16x32_bf16(false, A[c], false, B,
                                                    (short)0, acc, false, false);
    }

    float y[8];
#pragma unroll
    for (int r = 0; r < 8; ++r) {
      y[r] = acc[r] + bv[r];
      if (valid) { ls[r] += y[r]; lq[r] += y[r] * y[r]; }
    }
    if (valid) {
      float* zp = z + (size_t)s * 16 + f0;
      *(float4*)(zp)     = float4{y[0], y[1], y[2], y[3]};
      *(float4*)(zp + 4) = float4{y[4], y[5], y[6], y[7]};
    }
  }

#pragma unroll
  for (int r = 0; r < 8; ++r) {
    float s = ls[r], q = lq[r];
#pragma unroll
    for (int off = 8; off >= 1; off >>= 1) {
      s += __shfl_xor(s, off, 16);
      q += __shfl_xor(q, off, 16);
    }
    if ((lane & 15) == 0) {
      atomicAdd(&statsOut[f0 + r], s);
      atomicAdd(&statsOut[16 + f0 + r], q);
    }
  }
}

// ---------------------------------------------------------------------------
__global__ void embed_kernel(const float* __restrict__ x, const float* __restrict__ w1,
                             const float* __restrict__ b1, const float* __restrict__ w2,
                             const float* __restrict__ b2, float* __restrict__ nh, int Nn)
{
  int i = blockIdx.x * blockDim.x + threadIdx.x;
  if (i >= Nn * 8) return;
  int n = i >> 3, c = i & 7;
  float xv = x[n * 8 + c];
  float t[8];
#pragma unroll
  for (int mm = 0; mm < 8; ++mm) t[mm] = fmaxf(xv * w1[mm] + b1[mm], 0.f);
#pragma unroll
  for (int l = 0; l < 8; ++l) {
    float a = b2[l];
#pragma unroll
    for (int mm = 0; mm < 8; ++mm) a += t[mm] * w2[mm * 8 + l];
    nh[(size_t)n * 64 + c * 8 + l] = a;
  }
}

__global__ void head_kernel(const float* __restrict__ z, const float* __restrict__ stats,
                            const float* __restrict__ g, const float* __restrict__ be,
                            const float* __restrict__ w2, const float* __restrict__ b2,
                            const int* __restrict__ batch, float* gsum, float* gcnt,
                            float invN, int Ns)
{
  int nidx = blockIdx.x * blockDim.x + threadIdx.x;
  if (nidx >= Ns) return;
  float o = b2[0];
#pragma unroll
  for (int f = 0; f < 16; ++f) {
    float mu  = stats[f] * invN;
    float var = stats[16 + f] * invN - mu * mu;
    float v = fmaxf((z[(size_t)nidx * 16 + f] - mu) * rsqrtf(var + EPS_BN) * g[f] + be[f], 0.f);
    o += v * w2[f];
  }
  int bb = batch[nidx];
  atomicAdd(&gsum[bb], o);
  atomicAdd(&gcnt[bb], 1.f);
}

__global__ void finalize_kernel(const float* __restrict__ gsum,
                                const float* __restrict__ gcnt, float* __restrict__ out)
{
  int i = threadIdx.x;
  if (i < 32) out[i] = gsum[i] / fmaxf(gcnt[i], 1.f);
}

// ---------------------------------------------------------------------------
extern "C" void kernel_launch(void* const* d_in, const int* in_sizes, int n_in,
                              void* d_out, int out_size, void* d_ws, size_t ws_size,
                              hipStream_t stream)
{
  const float* x     = (const float*)d_in[0];
  const int*   eidx  = (const int*)d_in[2];
  const int*   batch = (const int*)d_in[3];
  const int N = in_sizes[0] / 8;
  const int E = in_sizes[2] / 2;
  const int* rowI = eidx;       // edge_index[0] = source
  const int* colI = eidx + E;   // edge_index[1] = target

  int p = 4;
  const float* emb_w1 = (const float*)d_in[p++];
  const float* emb_b1 = (const float*)d_in[p++];
  const float* emb_w2 = (const float*)d_in[p++];
  const float* emb_b2 = (const float*)d_in[p++];
  struct RB { const float *w1,*b1,*g1,*be1,*w2,*b2,*g2,*be2,*ws,*bs; };
  RB rb[2][4];  // layer x {msg1, msg2, upd1, upd2}
  for (int L = 0; L < 2; ++L)
    for (int r = 0; r < 4; ++r) {
      RB& q = rb[L][r];
      q.w1 =(const float*)d_in[p++]; q.b1 =(const float*)d_in[p++];
      q.g1 =(const float*)d_in[p++]; q.be1=(const float*)d_in[p++];
      q.w2 =(const float*)d_in[p++]; q.b2 =(const float*)d_in[p++];
      q.g2 =(const float*)d_in[p++]; q.be2=(const float*)d_in[p++];
      q.ws =(const float*)d_in[p++]; q.bs =(const float*)d_in[p++];
    }
  const float* fc1_w = (const float*)d_in[p++];
  const float* fc1_b = (const float*)d_in[p++];
  const float* fc_g  = (const float*)d_in[p++];
  const float* fc_be = (const float*)d_in[p++];
  const float* fc2_w = (const float*)d_in[p++];
  const float* fc2_b = (const float*)d_in[p++];
  (void)n_in; (void)out_size; (void)ws_size;

  float* W = (float*)d_ws;
  size_t off = 0;
  float* eB1  = W + off; off += (size_t)E * 64;
  float* eB2  = W + off; off += (size_t)E * 64;
  float* nh   = W + off; off += (size_t)N * 64;
  float* nAgg = W + off; off += (size_t)N * 64;
  float* nB1  = W + off; off += (size_t)N * 64;
  float* nB2  = W + off; off += (size_t)N * 64;
  float* z    = W + off; off += (size_t)N * 16;
  float* gsum = W + off; off += 32;
  float* gcnt = W + off; off += 32;
  float* statsBase = W + off; off += 17 * 64;

  hipMemsetAsync(gsum, 0, (64 + 17 * 64) * sizeof(float), stream);

  embed_kernel<<<(N * 8 + 255) / 256, 256, 0, stream>>>(x, emb_w1, emb_b1, emb_w2, emb_b2, nh, N);

  const float invE8 = 1.f / ((float)E * 8.f);
  const float invN8 = 1.f / ((float)N * 8.f);
  int slot = 0;
  auto st = [&](int s) { return statsBase + s * 64; };
  dim3 cgrid(512), cblk(256);

  for (int L = 0; L < 2; ++L) {
    const RB& m1 = rb[L][0]; const RB& m2 = rb[L][1];
    const RB& u1 = rb[L][2]; const RB& u2 = rb[L][3];
    hipMemsetAsync(nAgg, 0, (size_t)N * 64 * sizeof(float), stream);

    // ---- message path (edges), resblock 1: X = gather concat ----
    float* s0 = st(slot++); float* s1 = st(slot++);
    conv_rb_kernel<16,1,false,true><<<cgrid, cblk, 0, stream>>>(
        nullptr, nh, nullptr, colI, rowI,
        m1.w1, m1.b1, m1.ws, m1.bs,
        nullptr, nullptr, nullptr, 0.f, eB1, eB2, s0, E);
    conv_rb_kernel<8,0,true,false><<<cgrid, cblk, 0, stream>>>(
        eB1, nullptr, nullptr, nullptr, nullptr,
        m1.w2, m1.b2, nullptr, nullptr,
        s0, m1.g1, m1.be1, invE8, eB1, nullptr, s1, E);
    combine_kernel<<<(unsigned)(((size_t)E * 16 + 255) / 256), 256, 0, stream>>>(
        eB1, eB2, s1, m1.g2, m1.be2, invE8, eB1, nullptr, nullptr, E);

    // ---- message path, resblock 2 + scatter into nAgg ----
    s0 = st(slot++); s1 = st(slot++);
    conv_rb_kernel<8,0,false,true><<<cgrid, cblk, 0, stream>>>(
        eB1, nullptr, nullptr, nullptr, nullptr,
        m2.w1, m2.b1, m2.ws, m2.bs,
        nullptr, nullptr, nullptr, 0.f, eB2, eB1, s0, E);
    conv_rb_kernel<8,0,true,false><<<cgrid, cblk, 0, stream>>>(
        eB2, nullptr, nullptr, nullptr, nullptr,
        m2.w2, m2.b2, nullptr, nullptr,
        s0, m2.g1, m2.be1, invE8, eB2, nullptr, s1, E);
    combine_kernel<<<(unsigned)(((size_t)E * 16 + 255) / 256), 256, 0, stream>>>(
        eB2, eB1, s1, m2.g2, m2.be2, invE8, nullptr, colI, nAgg, E);

    // ---- update path (nodes), resblock 1: X = concat(h, agg) ----
    s0 = st(slot++); s1 = st(slot++);
    conv_rb_kernel<16,2,false,true><<<cgrid, cblk, 0, stream>>>(
        nullptr, nh, nAgg, nullptr, nullptr,
        u1.w1, u1.b1, u1.ws, u1.bs,
        nullptr, nullptr, nullptr, 0.f, nB1, nB2, s0, N);
    conv_rb_kernel<8,0,true,false><<<cgrid, cblk, 0, stream>>>(
        nB1, nullptr, nullptr, nullptr, nullptr,
        u1.w2, u1.b2, nullptr, nullptr,
        s0, u1.g1, u1.be1, invN8, nB1, nullptr, s1, N);
    combine_kernel<<<(unsigned)(((size_t)N * 16 + 255) / 256), 256, 0, stream>>>(
        nB1, nB2, s1, u1.g2, u1.be2, invN8, nB1, nullptr, nullptr, N);

    // ---- update path, resblock 2 -> new h ----
    s0 = st(slot++); s1 = st(slot++);
    conv_rb_kernel<8,0,false,true><<<cgrid, cblk, 0, stream>>>(
        nB1, nullptr, nullptr, nullptr, nullptr,
        u2.w1, u2.b1, u2.ws, u2.bs,
        nullptr, nullptr, nullptr, 0.f, nB2, nB1, s0, N);
    conv_rb_kernel<8,0,true,false><<<cgrid, cblk, 0, stream>>>(
        nB2, nullptr, nullptr, nullptr, nullptr,
        u2.w2, u2.b2, nullptr, nullptr,
        s0, u2.g1, u2.be1, invN8, nB2, nullptr, s1, N);
    combine_kernel<<<(unsigned)(((size_t)N * 16 + 255) / 256), 256, 0, stream>>>(
        nB2, nB1, s1, u2.g2, u2.be2, invN8, nh, nullptr, nullptr, N);
  }

  // ---- FC head ----
  float* sf = st(slot++);
  fc1_wmma_kernel<<<dim3(128), cblk, 0, stream>>>(nh, fc1_w, fc1_b, z, sf, N);
  head_kernel<<<(N + 255) / 256, 256, 0, stream>>>(z, sf, fc_g, fc_be, fc2_w, fc2_b,
                                                   batch, gsum, gcnt, 1.f / (float)N, N);
  finalize_kernel<<<1, 32, 0, stream>>>(gsum, gcnt, (float*)d_out);
}
